// MultiHeadAttention_84456236908630
// MI455X (gfx1250) — matrix operand order repaired and checked by reading, not compiled, not run on hardware
//
#include <hip/hip_runtime.h>
#include <cstdint>
#include <cstddef>

// ---------- CDNA5 WMMA types ----------
typedef __attribute__((ext_vector_type(16))) __bf16 v16bf;
typedef __attribute__((ext_vector_type(8)))  float  v8f;

union Frag {
  v16bf v;
  uint4 q[2];
};

__device__ __forceinline__ unsigned short f2bf(float f) {
  unsigned u = __float_as_uint(f);
  u += 0x7FFFu + ((u >> 16) & 1u);   // round-to-nearest-even
  return (unsigned short)(u >> 16);
}

__device__ __forceinline__ v8f wmma_bf16(const Frag& a, const Frag& b, v8f c) {
  return __builtin_amdgcn_wmma_f32_16x16x32_bf16(false, a.v, false, b.v,
                                                 (short)0, c, false, false);
}

__device__ __forceinline__ v8f zero8() {
  v8f z;
#pragma unroll
  for (int i = 0; i < 8; ++i) z[i] = 0.0f;
  return z;
}

// =====================================================================
// GEMM: C[4096,1024] = A[4096,1024] @ W[1024,1024]
// 128x128 tile per workgroup (256 threads = 8 waves), K-step 32.
// OUT_MODE 0: f32 row-major out
// OUT_MODE 1: bf16 per-head layout  [(b*16+h)*1024 + l]*64 + d   (Q,K)
// OUT_MODE 2: bf16 per-head transposed [(b*16+h)*64 + d]*1024 + l (V)
// =====================================================================
template <int OUT_MODE, bool A_BF16>
__global__ __launch_bounds__(256) void gemm128(const void* __restrict__ Ain,
                                               const float* __restrict__ W,
                                               void* __restrict__ Cout) {
  __shared__ unsigned short sA[128 * 32];   // [row][k]  bf16
  __shared__ unsigned short sW[128 * 32];   // [n][k]    bf16 (transposed)

  const int tid   = threadIdx.x;
  const int tileN = blockIdx.x;             // 0..7
  const int tileM = blockIdx.y;             // 0..31
  const int w     = tid >> 5;
  const int lane  = tid & 31;
  const int hlf   = lane >> 4;
  const int l15   = lane & 15;
  const int wrow  = (w >> 1) * 32;
  const int wcol  = (w & 1) * 64;

  v8f acc[2][4];
#pragma unroll
  for (int i = 0; i < 2; ++i)
#pragma unroll
    for (int j = 0; j < 4; ++j) acc[i][j] = zero8();

  const int arow  = tid >> 1;               // 0..127
  const int ahalf = tid & 1;                // 0/1 (16-elem half of k-tile)
  const int krow  = tid >> 3;               // 0..31
  const int nseg  = tid & 7;                // 0..7

  for (int kt = 0; kt < 1024; kt += 32) {
    __syncthreads();
    // ---- stage A tile (convert f32->bf16 unless already bf16) ----
    {
      unsigned short* dst = &sA[arow * 32 + ahalf * 16];
      if (A_BF16) {
        const unsigned short* pa = (const unsigned short*)Ain +
            (size_t)(tileM * 128 + arow) * 1024 + kt + ahalf * 16;
        ((uint4*)dst)[0] = ((const uint4*)pa)[0];
        ((uint4*)dst)[1] = ((const uint4*)pa)[1];
        if (kt + 32 < 1024) __builtin_prefetch(pa + 32, 0, 0);
      } else {
        const float* pa = (const float*)Ain +
            (size_t)(tileM * 128 + arow) * 1024 + kt + ahalf * 16;
#pragma unroll
        for (int c = 0; c < 4; ++c) {
          float4 f = ((const float4*)pa)[c];
          dst[c * 4 + 0] = f2bf(f.x);
          dst[c * 4 + 1] = f2bf(f.y);
          dst[c * 4 + 2] = f2bf(f.z);
          dst[c * 4 + 3] = f2bf(f.w);
        }
        if (kt + 32 < 1024) __builtin_prefetch(pa + 32, 0, 0);
      }
    }
    // ---- stage W tile transposed into LDS ----
    {
      const float* pw = W + (size_t)(kt + krow) * 1024 + tileN * 128 + nseg * 16;
#pragma unroll
      for (int c = 0; c < 4; ++c) {
        float4 f = ((const float4*)pw)[c];
        int nl = nseg * 16 + c * 4;
        sW[(nl + 0) * 32 + krow] = f2bf(f.x);
        sW[(nl + 1) * 32 + krow] = f2bf(f.y);
        sW[(nl + 2) * 32 + krow] = f2bf(f.z);
        sW[(nl + 3) * 32 + krow] = f2bf(f.w);
      }
      if (kt + 32 < 1024) __builtin_prefetch(pw + 32 * 1024, 0, 0);
    }
    __syncthreads();

    // ---- fragments + 8 WMMAs ----
    Frag a0, a1;
    {
      const unsigned short* p0 = &sA[(wrow + l15) * 32 + hlf * 8];
      a0.q[0] = *(const uint4*)p0;
      a0.q[1] = *(const uint4*)(p0 + 16);
      const unsigned short* p1 = &sA[(wrow + 16 + l15) * 32 + hlf * 8];
      a1.q[0] = *(const uint4*)p1;
      a1.q[1] = *(const uint4*)(p1 + 16);
    }
#pragma unroll
    for (int j = 0; j < 4; ++j) {
      Frag bf;
      const unsigned short* pb = &sW[(wcol + j * 16 + l15) * 32 + hlf * 8];
      bf.q[0] = *(const uint4*)pb;
      bf.q[1] = *(const uint4*)(pb + 16);
      acc[0][j] = wmma_bf16(a0, bf, acc[0][j]);
      acc[1][j] = wmma_bf16(a1, bf, acc[1][j]);
    }
  }

  // ---- epilogue ----
#pragma unroll
  for (int i = 0; i < 2; ++i)
#pragma unroll
    for (int j = 0; j < 4; ++j)
#pragma unroll
      for (int v = 0; v < 8; ++v) {
        int gm = tileM * 128 + wrow + i * 16 + hlf * 8 + v;  // row in [0,4096)
        int gn = tileN * 128 + wcol + j * 16 + l15;          // col in [0,1024)
        float val = acc[i][j][v];
        if (OUT_MODE == 0) {
          ((float*)Cout)[(size_t)gm * 1024 + gn] = val;
        } else if (OUT_MODE == 1) {
          size_t dst = ((size_t)((gm >> 10) * 16 + (gn >> 6)) << 16) +
                       (size_t)(((gm & 1023) << 6) + (gn & 63));
          ((unsigned short*)Cout)[dst] = f2bf(val);
        } else {
          size_t dst = ((size_t)((gm >> 10) * 16 + (gn >> 6)) << 16) +
                       (size_t)(((gn & 63) << 10) + (gm & 1023));
          ((unsigned short*)Cout)[dst] = f2bf(val);
        }
      }
}

// =====================================================================
// Flash attention with ESIM masked softmax.
// Grid: (Lq/64, H, B), 128 threads = 4 waves; each wave owns 16 q-rows.
// Qh,Kh: bf16 [(b*16+h)*1024 + l]*64 + d ; Vt: bf16 [(b*16+h)*64 + d]*1024 + l
// X out: bf16 [b*1024+q][h*64+d], query mask applied.
// =====================================================================
__global__ __launch_bounds__(128) void attn_fa(const unsigned short* __restrict__ Qh,
                                               const unsigned short* __restrict__ Kh,
                                               const unsigned short* __restrict__ Vt,
                                               const int* __restrict__ qmask,
                                               const int* __restrict__ kmask,
                                               unsigned short* __restrict__ X) {
  __shared__ unsigned short sP[4][16 * 32];   // wave-private P staging (transpose D->A)

  const int tid  = threadIdx.x;
  const int w    = tid >> 5;
  const int lane = tid & 31;
  const int hlf  = lane >> 4;
  const int l15  = lane & 15;
  const int b    = blockIdx.z;
  const int h    = blockIdx.y;
  const int qw   = blockIdx.x * 64 + w * 16;
  const size_t bh = (size_t)(b * 16 + h) * 65536;   // 1024*64 elems per (b,h)

  // Resident Q fragments (16 rows x 64 d => two 16x32 A-fragments)
  Frag qf[2];
  {
    const unsigned short* qp = Qh + bh + (size_t)(qw + l15) * 64;
#pragma unroll
    for (int df = 0; df < 2; ++df) {
      int kb = df * 32 + hlf * 8;
      qf[df].q[0] = *(const uint4*)(qp + kb);
      qf[df].q[1] = *(const uint4*)(qp + kb + 16);
    }
  }

  float m[8], l[8];
#pragma unroll
  for (int v = 0; v < 8; ++v) { m[v] = -__builtin_inff(); l[v] = 0.0f; }
  v8f o[4];
#pragma unroll
  for (int j = 0; j < 4; ++j) o[j] = zero8();

  const int* kmb = kmask + b * 1024;

  for (int k0 = 0; k0 < 1024; k0 += 32) {
    // ---- scores: two 16q x 16k tiles, 2 WMMAs each (d = 64 in two steps) ----
    v8f s0 = zero8(), s1 = zero8();
#pragma unroll
    for (int df = 0; df < 2; ++df) {
      Frag kb0, kb1;
      const unsigned short* kp = Kh + bh + (size_t)(k0 + l15) * 64 + df * 32 + hlf * 8;
      kb0.q[0] = *(const uint4*)kp;
      kb0.q[1] = *(const uint4*)(kp + 16);
      const unsigned short* kp1 = kp + 16 * 64;
      kb1.q[0] = *(const uint4*)kp1;
      kb1.q[1] = *(const uint4*)(kp1 + 16);
      s0 = wmma_bf16(qf[df], kb0, s0);
      s1 = wmma_bf16(qf[df], kb1, s1);
    }

    const int km0 = kmb[k0 + l15];
    const int km1 = kmb[k0 + 16 + l15];

    float p0[8], p1[8], mx[8], fac[8], rs[8];
#pragma unroll
    for (int v = 0; v < 8; ++v) {
      float a0 = km0 ? s0[v] * 0.125f : -__builtin_inff();
      float a1 = km1 ? s1[v] * 0.125f : -__builtin_inff();
      p0[v] = a0;
      p1[v] = a1;
      mx[v] = fmaxf(a0, a1);
    }
#pragma unroll
    for (int xm = 1; xm < 16; xm <<= 1)
#pragma unroll
      for (int v = 0; v < 8; ++v) mx[v] = fmaxf(mx[v], __shfl_xor(mx[v], xm, 32));

#pragma unroll
    for (int v = 0; v < 8; ++v) {
      float mnew  = fmaxf(m[v], mx[v]);
      float msafe = (mnew < -1e30f) ? 0.0f : mnew;
      float e0 = km0 ? __expf(p0[v] - msafe) : 0.0f;
      float e1 = km1 ? __expf(p1[v] - msafe) : 0.0f;
      fac[v] = __expf(m[v] - msafe);   // exp(-inf)=0 handles first tile
      m[v]  = mnew;
      p0[v] = e0;
      p1[v] = e1;
      rs[v] = e0 + e1;
    }
#pragma unroll
    for (int xm = 1; xm < 16; xm <<= 1)
#pragma unroll
      for (int v = 0; v < 8; ++v) rs[v] += __shfl_xor(rs[v], xm, 32);

#pragma unroll
    for (int v = 0; v < 8; ++v) l[v] = l[v] * fac[v] + rs[v];
#pragma unroll
    for (int j = 0; j < 4; ++j)
#pragma unroll
      for (int v = 0; v < 8; ++v) o[j][v] *= fac[v];

    // ---- transpose P (D-layout -> A-layout) via wave-private LDS ----
#pragma unroll
    for (int v = 0; v < 8; ++v) {
      int r = hlf * 8 + v;
      sP[w][r * 32 + l15]      = f2bf(p0[v]);
      sP[w][r * 32 + 16 + l15] = f2bf(p1[v]);
    }
    asm volatile("s_wait_dscnt 0" ::: "memory");   // wave-local: DS counter only
    Frag pf;
    pf.q[0] = *(const uint4*)&sP[w][l15 * 32 + hlf * 8];
    pf.q[1] = *(const uint4*)&sP[w][l15 * 32 + hlf * 8 + 16];

    // ---- O += P(16x32) @ V(32x64), 4 WMMAs, B-frags straight from global Vt ----
#pragma unroll
    for (int j = 0; j < 4; ++j) {
      Frag vb;
      const unsigned short* vp =
          Vt + bh + (size_t)(j * 16 + l15) * 1024 + k0 + hlf * 8;
      vb.q[0] = *(const uint4*)vp;
      vb.q[1] = *(const uint4*)(vp + 16);
      o[j] = wmma_bf16(pf, vb, o[j]);
    }
  }

  // ---- epilogue: renormalize (ESIM eps), apply query mask, store bf16 X ----
  const int* qmb = qmask + b * 1024;
  float inv[8];
#pragma unroll
  for (int v = 0; v < 8; ++v) {
    int r = qw + hlf * 8 + v;
    float qm = qmb[r] ? 1.0f : 0.0f;
    inv[v] = qm / (l[v] + 1e-13f);
  }
#pragma unroll
  for (int j = 0; j < 4; ++j)
#pragma unroll
    for (int v = 0; v < 8; ++v) {
      int r = qw + hlf * 8 + v;
      size_t dst = (size_t)(b * 1024 + r) * 1024 + h * 64 + j * 16 + l15;
      X[dst] = f2bf(o[j][v] * inv[v]);
    }
}

// =====================================================================
extern "C" void kernel_launch(void* const* d_in, const int* in_sizes, int n_in,
                              void* d_out, int out_size, void* d_ws, size_t ws_size,
                              hipStream_t stream) {
  (void)in_sizes; (void)n_in; (void)out_size; (void)ws_size;
  const float* q     = (const float*)d_in[0];
  const float* k     = (const float*)d_in[1];
  const float* v     = (const float*)d_in[2];
  const int*   qmask = (const int*)d_in[3];
  const int*   kmask = (const int*)d_in[4];
  const float* Wq    = (const float*)d_in[5];
  const float* Wk    = (const float*)d_in[6];
  const float* Wv    = (const float*)d_in[7];
  const float* Wo    = (const float*)d_in[8];

  char* ws = (char*)d_ws;
  unsigned short* Qh = (unsigned short*)(ws);                              // 8 MB
  unsigned short* Kh = (unsigned short*)(ws + (size_t)8  * 1024 * 1024);   // 8 MB
  unsigned short* Vt = (unsigned short*)(ws + (size_t)16 * 1024 * 1024);   // 8 MB
  unsigned short* X  = (unsigned short*)(ws + (size_t)24 * 1024 * 1024);   // 8 MB

  dim3 gg(8, 32), gb(256);
  gemm128<1, false><<<gg, gb, 0, stream>>>(q, Wq, Qh);
  gemm128<1, false><<<gg, gb, 0, stream>>>(k, Wk, Kh);
  gemm128<2, false><<<gg, gb, 0, stream>>>(v, Wv, Vt);
  attn_fa<<<dim3(16, 16, 4), 128, 0, stream>>>(Qh, Kh, Vt, qmask, kmask, X);
  gemm128<0, true><<<gg, gb, 0, stream>>>(X, Wo, d_out);
}
